// MultiheadAttention_1176821039368
// MI455X (gfx1250) — matrix operand order repaired and checked by reading, not compiled, not run on hardware
//
#include <hip/hip_runtime.h>
#include <hip/hip_bf16.h>

typedef __attribute__((ext_vector_type(16))) _Float16 v16h;
typedef __attribute__((ext_vector_type(8)))  float    v8f;
typedef __attribute__((ext_vector_type(4)))  unsigned u32x4;
typedef __attribute__((ext_vector_type(8)))  unsigned u32x8;

#define B_    2
#define S_    2048
#define H_    1024
#define NH_   16
#define HD_   64
#define MTOT  (B_ * S_)      // 4096
#define LPAD  40             // LDS row stride (halves) for manual fills
#define LPAD2 72             // LDS row stride (halves) for TDM-padded 64-wide rows

// ---------- WMMA fragment loads from LDS (layouts per ISA 7.12.2, wave32) ----------

// A 16x32 f16 (MxK): lane m = lane&15, kb = (lane>>4)*8.
// VGPR p (0..3): K = kb + 2p, +1 ; VGPR p (4..7): K = kb + 16 + 2(p-4), +1.
__device__ inline v16h frag_a_lds(const _Float16* base, int ldr, int lane) {
  const int m  = lane & 15;
  const int kb = (lane >> 4) << 3;
  union { v16h v; unsigned u[8]; } r;
  const _Float16* row = base + m * ldr;
#pragma unroll
  for (int p = 0; p < 8; ++p) {
    const int k = ((p >> 2) << 4) + kb + ((p & 3) << 1);
    r.u[p] = *(const unsigned*)(row + k);
  }
  return r.v;
}

// B 32x16 f16 (KxN), LDS stored [n][k] (k contiguous): lane n = lane&15,
// kb = (lane>>4)*8; VGPR p (0..3): K = kb + 2p, +1 ; VGPR p (4..7): K = kb + 16 + 2(p-4), +1.
__device__ inline v16h frag_b_lds(const _Float16* base, int ldr, int lane) {
  const int n  = lane & 15;
  const int kb = (lane >> 4) << 3;
  union { v16h v; unsigned u[8]; } r;
  const _Float16* row = base + n * ldr + kb;
#pragma unroll
  for (int p = 0; p < 8; ++p) r.u[p] = *(const unsigned*)(row + ((p >> 2) << 4) + ((p & 3) << 1));
  return r.v;
}

// LDS byte offset of a __shared__ pointer (flat LDS-aperture addr: low 32 bits = offset)
__device__ inline unsigned lds_offset(const void* p) {
  return (unsigned)(size_t)p;
}

// ---------- Tensor Data Mover: 2D f16 tile Global -> LDS (ISA ch.8, D# groups 0/1) ----
// tile_d0 halves per row (row = tile_d0*2 bytes), tile_d1 rows, row stride stride0
// halves in memory. LDS padding: pad_interval code 4 => every 32 DWORDs (64 halves),
// pad_amount code 3 => 4 DWORDs (8 halves) -> effective LDS row stride 72 halves.
__device__ inline void tdm_load_2d_f16(const _Float16* gbase, unsigned lds_byte_off,
                                       unsigned tensor_d0, unsigned tensor_d1,
                                       unsigned tile_d0, unsigned tile_d1,
                                       unsigned stride0) {
  const unsigned long long ga = (unsigned long long)(size_t)gbase;
  u32x4 g0;
  g0[0] = 0x1u;                                   // count=1, user descriptor
  g0[1] = lds_byte_off;                           // lds_addr
  g0[2] = (unsigned)ga;                           // global_addr[31:0]
  g0[3] = ((unsigned)(ga >> 32) & 0x01FFFFFFu)    // global_addr[56:32]
          | 0x80000000u;                          // type=2 ("image")
  u32x8 g1;
  g1[0] = (1u << 16)                              // data_size = 2 bytes
        | (1u << 20)                              // pad_enable
        | (4u << 22)                              // pad_interval: 32 DWORDs
        | (3u << 25);                             // pad_amount : 4 DWORDs
  g1[1] = (tensor_d0 & 0xFFFFu) << 16;            // tensor_dim0[15:0]
  g1[2] = ((tensor_d0 >> 16) & 0xFFFFu) | ((tensor_d1 & 0xFFFFu) << 16);
  g1[3] = ((tensor_d1 >> 16) & 0xFFFFu) | ((tile_d0 & 0xFFFFu) << 16);
  g1[4] = (tile_d1 & 0xFFFFu);                    // tile_dim1; tile_dim2 = 0
  g1[5] = stride0;                                // tensor_dim0_stride[31:0]
  g1[6] = 0u;
  g1[7] = 0u;
  asm volatile("tensor_load_to_lds %0, %1" :: "s"(g0), "s"(g1) : "memory");
}

// ---------- Generic 128x128 tile GEMM: C = A(MxK) * W(NxK)^T + bias ----------
__global__ __launch_bounds__(256)
void proj_gemm_kernel(const float* __restrict__ A32, const _Float16* __restrict__ A16,
                      const float* __restrict__ W, const float* __restrict__ bias,
                      _Float16* __restrict__ out_hs, float* __restrict__ out_rm, int K) {
  __shared__ _Float16 lsA[128 * LPAD];
  __shared__ _Float16 lsB[128 * LPAD];
  const int tid  = threadIdx.x;
  const int lane = tid & 31;
  const int wv   = tid >> 5;            // 0..7
  const int wm   = (wv & 1) << 6;       // wave m-offset: 0 or 64
  const int wn   = (wv >> 1) << 5;      // wave n-offset: 0,32,64,96
  const int n0   = blockIdx.x * 128;
  const int m0   = blockIdx.y * 128;

  v8f acc[4][2] = {};

  const int row = tid >> 1;             // 0..127
  const int kh  = (tid & 1) << 4;       // 0 or 16

  for (int k0 = 0; k0 < K; k0 += 32) {
    if (A32) {
      const float* src = A32 + (size_t)(m0 + row) * K + k0 + kh;
#pragma unroll
      for (int e = 0; e < 16; ++e) lsA[row * LPAD + kh + e] = (_Float16)src[e];
    } else {
      const _Float16* src = A16 + (size_t)(m0 + row) * K + k0 + kh;
#pragma unroll
      for (int e = 0; e < 16; ++e) lsA[row * LPAD + kh + e] = src[e];
    }
    {
      const float* src = W + (size_t)(n0 + row) * K + k0 + kh;
#pragma unroll
      for (int e = 0; e < 16; ++e) lsB[row * LPAD + kh + e] = (_Float16)src[e];
    }
    __syncthreads();

    v16h bf[2];
#pragma unroll
    for (int ni = 0; ni < 2; ++ni)
      bf[ni] = frag_b_lds(lsB + (wn + ni * 16) * LPAD, LPAD, lane);
#pragma unroll
    for (int mi = 0; mi < 4; ++mi) {
      v16h af = frag_a_lds(lsA + (wm + mi * 16) * LPAD, LPAD, lane);
#pragma unroll
      for (int ni = 0; ni < 2; ++ni)
        acc[mi][ni] = __builtin_amdgcn_wmma_f32_16x16x32_f16(
            false, af, false, bf[ni], (short)0, acc[mi][ni], false, false);
    }
    __syncthreads();
  }

  // C/D layout: n = lane&15, m = p + (lane>>4)*8
  const int cn   = lane & 15;
  const int mofs = (lane >> 4) << 3;
#pragma unroll
  for (int mi = 0; mi < 4; ++mi)
#pragma unroll
    for (int ni = 0; ni < 2; ++ni)
#pragma unroll
      for (int p = 0; p < 8; ++p) {
        const int m = m0 + wm + mi * 16 + mofs + p;
        const int n = n0 + wn + ni * 16 + cn;
        const float val = acc[mi][ni][p] + bias[n];
        if (out_rm) out_rm[(size_t)m * H_ + n] = val;
        if (out_hs) {
          const int b = m >> 11, s = m & (S_ - 1);
          const int h = n >> 6,  d = n & (HD_ - 1);
          out_hs[(((size_t)(b * NH_ + h)) * S_ + s) * HD_ + d] = (_Float16)val;
        }
      }
}

// ---------- scores[b,h] = scale * Qh @ Kh^T  (2048x2048x64), TDM-fed tiles ----------
__global__ __launch_bounds__(256)
void scores_kernel(const _Float16* __restrict__ qh, const _Float16* __restrict__ khp,
                   float* __restrict__ attn, float scale) {
  __shared__ _Float16 lsA[128 * LPAD2];   // Q tile 128 x 64 (+pad)
  __shared__ _Float16 lsB[128 * LPAD2];   // K tile 128 x 64 (+pad)
  const int bh = blockIdx.z;
  const _Float16* Q  = qh  + (size_t)bh * S_ * HD_;
  const _Float16* Kh = khp + (size_t)bh * S_ * HD_;
  float* out = attn + (size_t)bh * S_ * S_;

  const int tid  = threadIdx.x;
  const int lane = tid & 31;
  const int wv   = tid >> 5;
  const int wm   = (wv & 1) << 6;
  const int wn   = (wv >> 1) << 5;
  const int n0   = blockIdx.x * 128;
  const int m0   = blockIdx.y * 128;

  // One wave DMAs both whole tiles via the Tensor Data Mover, then the
  // workgroup barrier publishes LDS to the other waves.
  if (wv == 0) {
    tdm_load_2d_f16(Q  + (size_t)m0 * HD_, lds_offset(lsA), HD_, S_, HD_, 128, HD_);
    tdm_load_2d_f16(Kh + (size_t)n0 * HD_, lds_offset(lsB), HD_, S_, HD_, 128, HD_);
    __builtin_amdgcn_s_wait_tensorcnt(0);
  }
  __syncthreads();

  v8f acc[4][2] = {};
#pragma unroll
  for (int k0 = 0; k0 < HD_; k0 += 32) {
    v16h bf[2];
#pragma unroll
    for (int ni = 0; ni < 2; ++ni)
      bf[ni] = frag_b_lds(lsB + (wn + ni * 16) * LPAD2 + k0, LPAD2, lane);
#pragma unroll
    for (int mi = 0; mi < 4; ++mi) {
      v16h af = frag_a_lds(lsA + (wm + mi * 16) * LPAD2 + k0, LPAD2, lane);
#pragma unroll
      for (int ni = 0; ni < 2; ++ni)
        acc[mi][ni] = __builtin_amdgcn_wmma_f32_16x16x32_f16(
            false, af, false, bf[ni], (short)0, acc[mi][ni], false, false);
    }
  }

  const int cn   = lane & 15;
  const int mofs = (lane >> 4) << 3;
#pragma unroll
  for (int mi = 0; mi < 4; ++mi)
#pragma unroll
    for (int ni = 0; ni < 2; ++ni)
#pragma unroll
      for (int p = 0; p < 8; ++p) {
        const int m = m0 + wm + mi * 16 + mofs + p;
        const int n = n0 + wn + ni * 16 + cn;
        out[(size_t)m * S_ + n] = acc[mi][ni][p] * scale;
      }
}

// ---------- row softmax in place over attn rows of length 2048 ----------
__global__ __launch_bounds__(256)
void softmax_kernel(float* __restrict__ attn) {
  __shared__ float red[256];
  const size_t rowi = blockIdx.x;                  // B*NH*S rows
  float* rowp = attn + rowi * (size_t)S_;
  const int tid = threadIdx.x;

  float vals[8];
  float mx = -__builtin_inff();
#pragma unroll
  for (int e = 0; e < 8; ++e) { vals[e] = rowp[tid + e * 256]; mx = fmaxf(mx, vals[e]); }
  red[tid] = mx; __syncthreads();
#pragma unroll
  for (int s = 128; s > 0; s >>= 1) {
    if (tid < s) red[tid] = fmaxf(red[tid], red[tid + s]);
    __syncthreads();
  }
  mx = red[0]; __syncthreads();

  float sum = 0.f;
#pragma unroll
  for (int e = 0; e < 8; ++e) { vals[e] = __expf(vals[e] - mx); sum += vals[e]; }
  red[tid] = sum; __syncthreads();
#pragma unroll
  for (int s = 128; s > 0; s >>= 1) {
    if (tid < s) red[tid] += red[tid + s];
    __syncthreads();
  }
  const float inv = 1.0f / red[0];
#pragma unroll
  for (int e = 0; e < 8; ++e) rowp[tid + e * 256] = vals[e] * inv;
}

// ---------- ctx[b,h] = attn @ Vh  (2048x64x2048), out f16 [b][s][h][d] ----------
__global__ __launch_bounds__(256)
void av_kernel(const float* __restrict__ attn, const _Float16* __restrict__ vh,
               _Float16* __restrict__ ao) {
  __shared__ _Float16 lsA[128 * LPAD];
  __shared__ _Float16 lsB[64 * LPAD];      // V tile stored transposed [d][k]
  const int bh = blockIdx.y;
  const int m0 = blockIdx.x * 128;
  const float* A = attn + (size_t)bh * S_ * S_;
  const _Float16* V = vh + (size_t)bh * S_ * HD_;
  const int b = bh >> 4, h = bh & 15;

  const int tid  = threadIdx.x;
  const int lane = tid & 31;
  const int wv   = tid >> 5;               // wave -> 16-row m strip

  v8f acc[4] = {};
  const int row = tid >> 1;
  const int kh  = (tid & 1) << 4;

  for (int k0 = 0; k0 < S_; k0 += 32) {
    {
      const float* src = A + (size_t)(m0 + row) * S_ + k0 + kh;
#pragma unroll
      for (int e = 0; e < 16; ++e) lsA[row * LPAD + kh + e] = (_Float16)src[e];
      // prefetch next attn tile (bandwidth-critical 536MB stream)
      __builtin_prefetch(src + ((k0 + 32 < S_) ? 32 : 0), 0, 1);
    }
#pragma unroll
    for (int e = 0; e < 8; ++e) {          // 32x64 V tile, transposed into LDS
      const int idx = tid * 8 + e;
      const int k = idx >> 6, d = idx & 63;
      lsB[d * LPAD + k] = V[(size_t)(k0 + k) * HD_ + d];
    }
    __syncthreads();

    v16h af = frag_a_lds(lsA + (wv * 16) * LPAD, LPAD, lane);
#pragma unroll
    for (int ni = 0; ni < 4; ++ni) {
      v16h bf = frag_b_lds(lsB + (ni * 16) * LPAD, LPAD, lane);
      acc[ni] = __builtin_amdgcn_wmma_f32_16x16x32_f16(
          false, af, false, bf, (short)0, acc[ni], false, false);
    }
    __syncthreads();
  }

  const int cn   = lane & 15;
  const int mofs = (lane >> 4) << 3;
#pragma unroll
  for (int ni = 0; ni < 4; ++ni)
#pragma unroll
    for (int p = 0; p < 8; ++p) {
      const int m = m0 + wv * 16 + mofs + p;       // s_q
      const int d = ni * 16 + cn;
      ao[((size_t)(b * S_ + m) * NH_ + h) * HD_ + d] = (_Float16)acc[ni][p];
    }
}

extern "C" void kernel_launch(void* const* d_in, const int* in_sizes, int n_in,
                              void* d_out, int out_size, void* d_ws, size_t ws_size,
                              hipStream_t stream) {
  const float* q    = (const float*)d_in[0];
  const float* k    = (const float*)d_in[1];
  const float* v    = (const float*)d_in[2];
  const float* wq_w = (const float*)d_in[3];
  const float* wq_b = (const float*)d_in[4];
  const float* wk_w = (const float*)d_in[5];
  const float* wk_b = (const float*)d_in[6];
  const float* wv_w = (const float*)d_in[7];
  const float* wv_b = (const float*)d_in[8];
  const float* wo_w = (const float*)d_in[9];
  const float* wo_b = (const float*)d_in[10];

  float* final_out = (float*)d_out;                         // [B,S,H] fp32
  float* attn      = (float*)d_out + (size_t)B_ * S_ * H_;  // [B,NH,S,S] fp32

  // workspace: Qh/Kh/Vh head-split f16 (8MB each) + ctx f16 (8MB) = 32MB
  _Float16* qh = (_Float16*)d_ws;
  _Float16* kh = qh + (size_t)MTOT * H_;
  _Float16* vh = kh + (size_t)MTOT * H_;
  _Float16* ao = vh + (size_t)MTOT * H_;

  const dim3 blk(256);
  const dim3 gProj(H_ / 128, MTOT / 128);
  proj_gemm_kernel<<<gProj, blk, 0, stream>>>(q, nullptr, wq_w, wq_b, qh, nullptr, H_);
  proj_gemm_kernel<<<gProj, blk, 0, stream>>>(k, nullptr, wk_w, wk_b, kh, nullptr, H_);
  proj_gemm_kernel<<<gProj, blk, 0, stream>>>(v, nullptr, wv_w, wv_b, vh, nullptr, H_);

  const dim3 gSc(S_ / 128, S_ / 128, B_ * NH_);
  scores_kernel<<<gSc, blk, 0, stream>>>(qh, kh, attn, 0.125f);   // 1/sqrt(64)

  softmax_kernel<<<dim3(B_ * NH_ * S_), blk, 0, stream>>>(attn);

  const dim3 gAV(S_ / 128, B_ * NH_);
  av_kernel<<<gAV, blk, 0, stream>>>(attn, vh, ao);

  proj_gemm_kernel<<<gProj, blk, 0, stream>>>(nullptr, ao, wo_w, wo_b, nullptr, final_out, H_);
}
